// NoteStateSequenceModule_89893665505923
// MI455X (gfx1250) — compile-verified, matrix operand order checked by reading, not run on hardware
//
#include <hip/hip_runtime.h>
#include <hip/hip_bf16.h>

typedef __attribute__((ext_vector_type(16))) __bf16 v16bf;
typedef __attribute__((ext_vector_type(8)))  float  v8f;

#define CFENCE() asm volatile("" ::: "memory")

// ---------------- LDS layout (bytes) ----------------
#define L_W1IH   0        // 192x192 bf16 fragments  (73728)
#define L_W1HH   73728    // 192x64  bf16 fragments  (24576)
#define L_W2IH   98304    // 192x64                  (24576)
#define L_W2HH   122880   // 192x64                  (24576)
#define L_BIAS1  147456   // 192 f32
#define L_BIAS2  148224   // 192 f32
#define L_FCW    148992   // 5x48 f32
#define L_FCB    149952   // 5 f32 (pad 64)
#define L_XA     150016   // 4 waves * 3072 bf16  (24576)
#define L_H1A    174592   // 4 waves * 1024 bf16  (8192)
#define L_H2A    182784   // 4 waves * 1024 bf16  (8192)
#define L_H2ROW  190976   // 4 waves * 16*48 f32  (12288)
#define L_TOTAL  203264

// ---------------- workspace layout (bytes) ----------------
#define WS_CTX   0         // 16*88*4 f32
#define WS_W1IH  32768
#define WS_W1HH  106496
#define WS_W2IH  131072
#define WS_W2HH  155648
#define WS_B1    180224
#define WS_B2    181248
#define WS_FCW   182272
#define WS_FCB   183296

__device__ __forceinline__ float sigf(float x) { return 1.0f / (1.0f + __expf(-x)); }
__device__ __forceinline__ unsigned short f2bf(float v) {
  __bf16 t = (__bf16)v;
  return __builtin_bit_cast(unsigned short, t);
}

// -------- context MLP: (B*P) rows of 3->16->16->4 --------
__global__ void ctx_net_kernel(const float* __restrict__ pc,
                               const float* __restrict__ w1, const float* __restrict__ b1,
                               const float* __restrict__ w2, const float* __restrict__ b2,
                               const float* __restrict__ w3, const float* __restrict__ b3,
                               float* __restrict__ ctx_out) {
  int idx = blockIdx.x * blockDim.x + threadIdx.x;
  if (idx >= 16 * 88) return;
  const float* x = pc + idx * 3;
  float h1[16], h2[16];
#pragma unroll
  for (int j = 0; j < 16; ++j) {
    float a = b1[j];
#pragma unroll
    for (int k = 0; k < 3; ++k) a += w1[j * 3 + k] * x[k];
    h1[j] = fmaxf(a, 0.0f);
  }
#pragma unroll
  for (int j = 0; j < 16; ++j) {
    float a = b2[j];
#pragma unroll
    for (int k = 0; k < 16; ++k) a += w2[j * 16 + k] * h1[k];
    h2[j] = fmaxf(a, 0.0f);
  }
#pragma unroll
  for (int j = 0; j < 4; ++j) {
    float a = b3[j];
#pragma unroll
    for (int k = 0; k < 16; ++k) a += w3[j * 16 + k] * h2[k];
    ctx_out[idx * 4 + j] = a;
  }
}

// -------- pack fp32 (192 x K) row-major weight into bf16 WMMA B-fragments --------
// Fragment order: [kc][nt][lane(32)][j(16)] ; lane holds output-col n = nt*16+(lane&15),
// K-slice k = kc*32 + (lane>>4)*16 + j ; zero-padded for k >= K.
__global__ void pack_wb_kernel(const float* __restrict__ W, unsigned short* __restrict__ out,
                               int K, int nElem) {
  int idx = blockIdx.x * blockDim.x + threadIdx.x;
  if (idx >= nElem) return;
  int j = idx & 15;
  int lane = (idx >> 4) & 31;
  int blk = idx >> 9;               // kc*12 + nt   (G=192 -> NT=12)
  int kc = blk / 12;
  int nt = blk - kc * 12;
  int n = nt * 16 + (lane & 15);
  int k = kc * 32 + (lane >> 4) * 16 + j;
  float v = (k < K) ? W[n * K + k] : 0.0f;
  out[idx] = f2bf(v);
}

// -------- fold biases, copy small FC params --------
__global__ void prep_small_kernel(const float* b1i, const float* b1h,
                                  const float* b2i, const float* b2h,
                                  const float* fw, const float* fb,
                                  float* bias1, float* bias2, float* fw_o, float* fb_o) {
  int i = blockIdx.x * blockDim.x + threadIdx.x;
  if (i < 192) { bias1[i] = b1i[i] + b1h[i]; bias2[i] = b2i[i] + b2h[i]; }
  if (i < 240) fw_o[i] = fw[i];
  if (i < 5)   fb_o[i] = fb[i];
}

// -------- fused 2-layer LSTM scan over pitch axis + fc_states + softmax --------
__launch_bounds__(128, 1)
__global__ void note_lstm_fused(const float* __restrict__ pf, const float* __restrict__ ctxbuf,
                                const unsigned short* __restrict__ gw1ih,
                                const unsigned short* __restrict__ gw1hh,
                                const unsigned short* __restrict__ gw2ih,
                                const unsigned short* __restrict__ gw2hh,
                                const float* __restrict__ gbias1, const float* __restrict__ gbias2,
                                const float* __restrict__ gfcw, const float* __restrict__ gfcb,
                                float* __restrict__ out) {
  extern __shared__ char smem[];
  const int tid = threadIdx.x;

  // ---- stage packed weights + small params into LDS (coalesced 16B copies) ----
  {
    uint4* d; const uint4* s;
    d = (uint4*)(smem + L_W1IH); s = (const uint4*)gw1ih;
    for (int i = tid; i < 73728 / 16; i += 128) d[i] = s[i];
    d = (uint4*)(smem + L_W1HH); s = (const uint4*)gw1hh;
    for (int i = tid; i < 24576 / 16; i += 128) d[i] = s[i];
    d = (uint4*)(smem + L_W2IH); s = (const uint4*)gw2ih;
    for (int i = tid; i < 24576 / 16; i += 128) d[i] = s[i];
    d = (uint4*)(smem + L_W2HH); s = (const uint4*)gw2hh;
    for (int i = tid; i < 24576 / 16; i += 128) d[i] = s[i];
    float* b1 = (float*)(smem + L_BIAS1);
    float* b2 = (float*)(smem + L_BIAS2);
    for (int i = tid; i < 192; i += 128) { b1[i] = gbias1[i]; b2[i] = gbias2[i]; }
    float* fw = (float*)(smem + L_FCW);
    for (int i = tid; i < 240; i += 128) fw[i] = gfcw[i];
    if (tid < 5) ((float*)(smem + L_FCB))[tid] = gfcb[tid];
  }

  const int wave = tid >> 5;
  const int lane = tid & 31;
  const int lhalf = lane >> 4;
  const int lmod = lane & 15;
  const int wg = blockIdx.x;         // 64 WGs: 4 per batch row
  const int b = wg >> 2;
  const int t0 = (wg & 3) * 64 + wave * 16;   // wave owns t0..t0+15

  unsigned short* xA  = (unsigned short*)(smem + L_XA)  + wave * 3072;
  unsigned short* h1A = (unsigned short*)(smem + L_H1A) + wave * 1024;
  unsigned short* h2A = (unsigned short*)(smem + L_H2A) + wave * 1024;
  float* h2r = (float*)(smem + L_H2ROW) + wave * 768;   // 16 x 48
  const float* biasL1 = (const float*)(smem + L_BIAS1);
  const float* biasL2 = (const float*)(smem + L_BIAS2);
  const float* fcwL = (const float*)(smem + L_FCW);
  const float* fcbL = (const float*)(smem + L_FCB);
  const v16bf* W1  = (const v16bf*)(smem + L_W1IH);
  const v16bf* Wh1 = (const v16bf*)(smem + L_W1HH);
  const v16bf* W2  = (const v16bf*)(smem + L_W2IH);
  const v16bf* Wh2 = (const v16bf*)(smem + L_W2HH);
  const v16bf* xF  = (const v16bf*)xA;
  const v16bf* h1F = (const v16bf*)h1A;
  const v16bf* h2F = (const v16bf*)h2A;

  // zero h state (A-fragment layout, includes K-pad 48..63)
  for (int i = lane; i < 1024; i += 32) { h1A[i] = 0; h2A[i] = 0; }
  float c1[24], c2[24];
#pragma unroll
  for (int i = 0; i < 24; ++i) { c1[i] = 0.0f; c2[i] = 0.0f; }
  __syncthreads();

  for (int p = 0; p < 88; ++p) {
    // ---- stage x tile into A-fragments (coalesced along t, fp32->bf16) ----
    const float* pfp = pf + (size_t)(b * 88 + p) * 188 * 256;
#pragma unroll 2
    for (int kk = 0; kk < 94; ++kk) {
      int h = kk * 2 + lhalf;
      float v = pfp[(size_t)h * 256 + t0 + lmod];
      int a = ((h >> 5) << 9) + ((((h >> 4) & 1) * 16 + lmod) << 4) + (h & 15);
      xA[a] = f2bf(v);
    }
    if (lane < 16) {
      const float* cp = ctxbuf + (b * 88 + p) * 4;
#pragma unroll
      for (int j = 0; j < 4; ++j) {
        int k = 188 + j;
        int a = ((k >> 5) << 9) + ((((k >> 4) & 1) * 16 + lane) << 4) + (k & 15);
        xA[a] = f2bf(cp[j]);
      }
    }
    CFENCE();   // wave-private LDS regions; LDS ops are in-order per wave

    // ---- LSTM1 gates: x(16x192)@W1ih^T + h1(16x64pad)@W1hh^T + bias ----
    v8f acc[12];
#pragma unroll
    for (int nt = 0; nt < 12; ++nt) {
      float bz = biasL1[nt * 16 + lmod];
      v8f a;
#pragma unroll
      for (int r = 0; r < 8; ++r) a[r] = bz;
      acc[nt] = a;
    }
#pragma unroll
    for (int kc = 0; kc < 6; ++kc) {
      v16bf a = xF[kc * 32 + lane];
#pragma unroll
      for (int nt = 0; nt < 12; ++nt) {
        v16bf bb = W1[(kc * 12 + nt) * 32 + lane];
        acc[nt] = __builtin_amdgcn_wmma_f32_16x16x32_bf16(false, a, false, bb, (short)0, acc[nt], false, false);
      }
    }
#pragma unroll
    for (int kc = 0; kc < 2; ++kc) {
      v16bf a = h1F[kc * 32 + lane];
#pragma unroll
      for (int nt = 0; nt < 12; ++nt) {
        v16bf bb = Wh1[(kc * 12 + nt) * 32 + lane];
        acc[nt] = __builtin_amdgcn_wmma_f32_16x16x32_bf16(false, a, false, bb, (short)0, acc[nt], false, false);
      }
    }
    CFENCE();

    // ---- LSTM1 cell: i/f/g/o share lane+row across N-tile accumulators ----
#pragma unroll
    for (int j = 0; j < 3; ++j) {
#pragma unroll
      for (int r = 0; r < 8; ++r) {
        float ig = sigf(acc[j][r]);
        float fg = sigf(acc[j + 3][r]);
        float gg = tanhf(acc[j + 6][r]);
        float og = sigf(acc[j + 9][r]);
        float c = fg * c1[j * 8 + r] + ig * gg;
        c1[j * 8 + r] = c;
        float hv = og * tanhf(c);
        int m = r + 8 * lhalf;
        int u = j * 16 + lmod;
        int a = ((u >> 5) << 9) + ((((u >> 4) & 1) * 16 + m) << 4) + (u & 15);
        h1A[a] = f2bf(hv);
      }
    }
    CFENCE();

    // ---- LSTM2 gates: h1@W2ih^T + h2@W2hh^T + bias ----
    v8f ac2[12];
#pragma unroll
    for (int nt = 0; nt < 12; ++nt) {
      float bz = biasL2[nt * 16 + lmod];
      v8f a;
#pragma unroll
      for (int r = 0; r < 8; ++r) a[r] = bz;
      ac2[nt] = a;
    }
#pragma unroll
    for (int kc = 0; kc < 2; ++kc) {
      v16bf a = h1F[kc * 32 + lane];
#pragma unroll
      for (int nt = 0; nt < 12; ++nt) {
        v16bf bb = W2[(kc * 12 + nt) * 32 + lane];
        ac2[nt] = __builtin_amdgcn_wmma_f32_16x16x32_bf16(false, a, false, bb, (short)0, ac2[nt], false, false);
      }
    }
#pragma unroll
    for (int kc = 0; kc < 2; ++kc) {
      v16bf a = h2F[kc * 32 + lane];
#pragma unroll
      for (int nt = 0; nt < 12; ++nt) {
        v16bf bb = Wh2[(kc * 12 + nt) * 32 + lane];
        ac2[nt] = __builtin_amdgcn_wmma_f32_16x16x32_bf16(false, a, false, bb, (short)0, ac2[nt], false, false);
      }
    }
    CFENCE();

    // ---- LSTM2 cell; write h2 to A-fragments (recurrence) + row-major f32 (logits) ----
#pragma unroll
    for (int j = 0; j < 3; ++j) {
#pragma unroll
      for (int r = 0; r < 8; ++r) {
        float ig = sigf(ac2[j][r]);
        float fg = sigf(ac2[j + 3][r]);
        float gg = tanhf(ac2[j + 6][r]);
        float og = sigf(ac2[j + 9][r]);
        float c = fg * c2[j * 8 + r] + ig * gg;
        c2[j * 8 + r] = c;
        float hv = og * tanhf(c);
        int m = r + 8 * lhalf;
        int u = j * 16 + lmod;
        int a = ((u >> 5) << 9) + ((((u >> 4) & 1) * 16 + m) << 4) + (u & 15);
        h2A[a] = f2bf(hv);
        h2r[m * 48 + u] = hv;
      }
    }
    CFENCE();

    // ---- logits (48->5) + softmax over 5 states; lanes 0..15 = 16 t values ----
    if (lane < 16) {
      float lg[5];
#pragma unroll
      for (int s = 0; s < 5; ++s) {
        float a = fcbL[s];
#pragma unroll 8
        for (int u = 0; u < 48; ++u) a += h2r[lane * 48 + u] * fcwL[s * 48 + u];
        lg[s] = a;
      }
      float mx = lg[0];
#pragma unroll
      for (int s = 1; s < 5; ++s) mx = fmaxf(mx, lg[s]);
      float ex[5], sum = 0.0f;
#pragma unroll
      for (int s = 0; s < 5; ++s) { ex[s] = __expf(lg[s] - mx); sum += ex[s]; }
      float inv = 1.0f / sum;
      size_t ob = ((size_t)(b * 88 + p) * 5) * 256 + t0 + lane;
#pragma unroll
      for (int s = 0; s < 5; ++s) out[ob + (size_t)s * 256] = ex[s] * inv;
    }
    CFENCE();
  }
}

extern "C" void kernel_launch(void* const* d_in, const int* in_sizes, int n_in,
                              void* d_out, int out_size, void* d_ws, size_t ws_size,
                              hipStream_t stream) {
  (void)in_sizes; (void)n_in; (void)out_size; (void)ws_size;
  const float* pf   = (const float*)d_in[0];
  const float* pc   = (const float*)d_in[1];
  const float* fc1w = (const float*)d_in[2];
  const float* fc1b = (const float*)d_in[3];
  const float* fc2w = (const float*)d_in[4];
  const float* fc2b = (const float*)d_in[5];
  const float* fc3w = (const float*)d_in[6];
  const float* fc3b = (const float*)d_in[7];
  const float* w1ih = (const float*)d_in[8];
  const float* w1hh = (const float*)d_in[9];
  const float* b1i  = (const float*)d_in[10];
  const float* b1h  = (const float*)d_in[11];
  const float* w2ih = (const float*)d_in[12];
  const float* w2hh = (const float*)d_in[13];
  const float* b2i  = (const float*)d_in[14];
  const float* b2h  = (const float*)d_in[15];
  const float* fcw  = (const float*)d_in[16];
  const float* fcb  = (const float*)d_in[17];
  float* out = (float*)d_out;
  char* ws = (char*)d_ws;

  float* ctx            = (float*)(ws + WS_CTX);
  unsigned short* pw1ih = (unsigned short*)(ws + WS_W1IH);
  unsigned short* pw1hh = (unsigned short*)(ws + WS_W1HH);
  unsigned short* pw2ih = (unsigned short*)(ws + WS_W2IH);
  unsigned short* pw2hh = (unsigned short*)(ws + WS_W2HH);
  float* bias1 = (float*)(ws + WS_B1);
  float* bias2 = (float*)(ws + WS_B2);
  float* fcwS  = (float*)(ws + WS_FCW);
  float* fcbS  = (float*)(ws + WS_FCB);

  ctx_net_kernel<<<(16 * 88 + 127) / 128, 128, 0, stream>>>(pc, fc1w, fc1b, fc2w, fc2b, fc3w, fc3b, ctx);
  pack_wb_kernel<<<(36864 + 255) / 256, 256, 0, stream>>>(w1ih, pw1ih, 192, 36864);
  pack_wb_kernel<<<(12288 + 255) / 256, 256, 0, stream>>>(w1hh, pw1hh, 48, 12288);
  pack_wb_kernel<<<(12288 + 255) / 256, 256, 0, stream>>>(w2ih, pw2ih, 48, 12288);
  pack_wb_kernel<<<(12288 + 255) / 256, 256, 0, stream>>>(w2hh, pw2hh, 48, 12288);
  prep_small_kernel<<<2, 128, 0, stream>>>(b1i, b1h, b2i, b2h, fcw, fcb, bias1, bias2, fcwS, fcbS);
  note_lstm_fused<<<64, 128, L_TOTAL, stream>>>(pf, ctx, pw1ih, pw1hh, pw2ih, pw2hh,
                                                bias1, bias2, fcwS, fcbS, out);
}
